// MultiHeadAttention2D_53858889892000
// MI455X (gfx1250) — compile-verified, matrix operand order checked by reading, not run on hardware
//
#include <hip/hip_runtime.h>
#include <hip/hip_bf16.h>

// Shapes
#define Gg 64
#define Mm 128
#define Dd 256

typedef _Float16 h4  __attribute__((ext_vector_type(4)));
typedef _Float16 h8  __attribute__((ext_vector_type(8)));
typedef _Float16 h16 __attribute__((ext_vector_type(16)));
typedef float    f8  __attribute__((ext_vector_type(8)));

// ---- fragment loaders -------------------------------------------------------
// A/B operand layout (16x32 f16): lanes 0-15 hold row/col = lane with
// K = {k0..k0+7, k0+16..k0+23}; lanes 16-31 hold K = {k0+8..15, k0+24..31}.
// Caller passes p already offset by (k0 + hi*8); we read p[0..7] and p[16..23].
__device__ __forceinline__ h16 frag_f16(const _Float16* p) {
    h16 r;
    *(h8*)&r     = *(const h8*)p;
    ((h8*)&r)[1] = *(const h8*)(p + 16);
    return r;
}
__device__ __forceinline__ h16 frag_f32(const float* p) {
    float4 a0 = *(const float4*)(p);
    float4 a1 = *(const float4*)(p + 4);
    float4 b0 = *(const float4*)(p + 16);
    float4 b1 = *(const float4*)(p + 20);
    h16 r;
    r[0]=(_Float16)a0.x; r[1]=(_Float16)a0.y; r[2]=(_Float16)a0.z; r[3]=(_Float16)a0.w;
    r[4]=(_Float16)a1.x; r[5]=(_Float16)a1.y; r[6]=(_Float16)a1.z; r[7]=(_Float16)a1.w;
    r[8]=(_Float16)b0.x; r[9]=(_Float16)b0.y; r[10]=(_Float16)b0.z; r[11]=(_Float16)b0.w;
    r[12]=(_Float16)b1.x; r[13]=(_Float16)b1.y; r[14]=(_Float16)b1.z; r[15]=(_Float16)b1.w;
    return r;
}
#define WMMA(Af, Bf, C) __builtin_amdgcn_wmma_f32_16x16x32_f16(false, (Af), false, (Bf), (short)0, (C), false, false)

// ---- P0: transpose-convert W (256x256 f32) -> Wt (f16, Wt[d*256+k]=W[k*256+d])
__global__ __launch_bounds__(256) void wt_kernel(const float* __restrict__ W,
                                                 _Float16* __restrict__ Wt) {
    int id = blockIdx.x * 256 + threadIdx.x;   // 65536 total
    int d = id >> 8, kk = id & 255;
    Wt[id] = (_Float16)W[kk * 256 + d];
}

// ---- P1/P5: GEMM  out(8192 x 256) = rowmap(A) (f32) x WtT + bias
// mode 0: store f16 row-major   mode 1: store f16 transposed per-group (G,D,M)
// mode 2: store f32 row-major
__global__ __launch_bounds__(256) void proj_kernel(const float* __restrict__ A,
                                                   const _Float16* __restrict__ Wt,
                                                   const float* __restrict__ bias,
                                                   void* __restrict__ outp,
                                                   int mode, int rowAdj, int rowOff) {
    const int wave = threadIdx.x >> 5;
    const int lane = threadIdx.x & 31;
    const int l16  = lane & 15;
    const int hi   = lane >> 4;
    const int tileId = blockIdx.x * 8 + wave;   // 8192 tiles
    const int rt = tileId >> 4;                 // 512 row tiles
    const int ct = tileId & 15;                 // 16 col tiles

    const int n = rt * 16 + l16;
    const int g = n >> 7;
    const int srcRow = n + g * rowAdj + rowOff;
    const float*    ap = A  + (size_t)srcRow * Dd + hi * 8;
    const _Float16* wp = Wt + (size_t)(ct * 16 + l16) * Dd + hi * 8;

    f8 acc = {};
    for (int k0 = 0; k0 < Dd; k0 += 32) {
        h16 af = frag_f32(ap + k0);
        h16 bf = frag_f16(wp + k0);
        acc = WMMA(af, bf, acc);
    }
    const float bv = bias[ct * 16 + l16];
    #pragma unroll
    for (int v = 0; v < 8; ++v) {
        const int row = rt * 16 + hi * 8 + v;
        const int col = ct * 16 + l16;
        const float val = acc[v] + bv;
        if (mode == 0) {
            ((_Float16*)outp)[(size_t)row * Dd + col] = (_Float16)val;
        } else if (mode == 1) {
            const int gg = row >> 7, ii = row & 127;
            ((_Float16*)outp)[((size_t)gg * Dd + col) * Mm + ii] = (_Float16)val;
        } else {
            ((float*)outp)[(size_t)row * Dd + col] = val;
        }
    }
}

// ---- P2: per (a,b): E = exp(Q'[a]·K'[b]T/16 + mask[a]*(-1e9)); block partial sums
// K'[b] staged once per block in LDS (row pad 8 halfs -> 528B stride, no 16-way
// bank conflicts), removing the 8x per-wave redundant global re-reads.
__global__ __launch_bounds__(256) void scores_kernel(const _Float16* __restrict__ Qh,
                                                     const _Float16* __restrict__ Kh,
                                                     const float* __restrict__ mask,
                                                     float* __restrict__ E,
                                                     float* __restrict__ bsums) {
    __shared__ _Float16 Kb[128][264];   // 128 x (256 + 8 pad)
    __shared__ float wsum[8];
    const int a = blockIdx.x >> 6;
    const int b = blockIdx.x & 63;
    const int tid  = threadIdx.x;
    const int wave = tid >> 5;
    const int lane = tid & 31;
    const int l16  = lane & 15;
    const int hi   = lane >> 4;

    // cooperative 16B-chunk copy: 4096 chunks of K'[b] (128x256 f16)
    #pragma unroll
    for (int i = 0; i < 16; ++i) {
        const int ch  = tid + i * 256;
        const int row = ch >> 5;
        const int cc  = (ch & 31) * 8;
        *(h8*)&Kb[row][cc] = *(const h8*)(Kh + ((size_t)b * Mm + row) * Dd + cc);
    }
    __syncthreads();

    const _Float16* qp = Qh + ((size_t)a * Mm + wave * 16 + l16) * Dd + hi * 8;

    f8 acc[8] = {};
    for (int k0 = 0; k0 < Dd; k0 += 32) {
        h16 af = frag_f16(qp + k0);
        #pragma unroll
        for (int t = 0; t < 8; ++t) {
            h16 bf = frag_f16(&Kb[t * 16 + l16][k0 + hi * 8]);
            acc[t] = WMMA(af, bf, acc[t]);
        }
    }

    float lsum = 0.f;
    const size_t ebase = ((size_t)(a * Gg + b)) << 14;
    #pragma unroll
    for (int t = 0; t < 8; ++t) {
        #pragma unroll
        for (int v = 0; v < 8; ++v) {
            const int i = wave * 16 + hi * 8 + v;
            const int j = t * 16 + l16;
            const float logit = acc[t][v] * 0.0625f
                              + mask[((size_t)a << 14) + (i << 7) + j] * (-1.0e9f);
            const float e = __expf(logit);
            E[ebase + (i << 7) + j] = e;
            lsum += e;
        }
    }
    // deterministic block reduction
    #pragma unroll
    for (int off = 16; off > 0; off >>= 1) lsum += __shfl_xor(lsum, off, 32);
    if (lane == 0) wsum[wave] = lsum;
    __syncthreads();
    if (tid == 0) {
        float s = 0.f;
        #pragma unroll
        for (int w = 0; w < 8; ++w) s += wsum[w];
        bsums[blockIdx.x] = s;
    }
}

// ---- P2b: deterministic denominator reduce (fixed order), 128 softmax rows
__global__ __launch_bounds__(128) void denred_kernel(const float* __restrict__ bsums,
                                                     float* __restrict__ invden) {
    const int r = threadIdx.x;            // 0..127
    const int a = r >> 1;
    const int b0 = (r & 1) * 32;
    float s = 0.f;
    for (int q = 0; q < 32; ++q) s += bsums[a * Gg + b0 + q];
    invden[r] = 1.0f / s;
}

// ---- P3: OutPre[a,m,d] = sum_b (inv[r(a,b)]) * E[a,b,m,:] . Vs[b,:,d]
// Fuses att normalization: each E element is read exactly once (cooperative
// slab stage into LDS) and the normalized f32 value is written back in place.
__global__ __launch_bounds__(256) void av_kernel(float* __restrict__ E,
                                                 const _Float16* __restrict__ VsT,
                                                 const float* __restrict__ invden,
                                                 float* __restrict__ OutPre) {
    __shared__ _Float16 Eh[32][136];     // 32 x (128 + 8 pad)  unnormalized e, f16
    __shared__ _Float16 Vb[256][136];    // 256 x (128 + 8 pad) Vs[b] (d-major)
    const int a  = blockIdx.x >> 2;
    const int mb = blockIdx.x & 3;
    const int tid  = threadIdx.x;
    const int wave = tid >> 5;
    const int lane = tid & 31;
    const int l16  = lane & 15;
    const int hi   = lane >> 4;
    const int wm = wave >> 2;            // 0..1  (m sub-slab)
    const int wn = wave & 3;             // 0..3  (d quarter)
    const int m0 = wm * 16;              // within 32-row slab
    const int d0 = wn * 64;
    const int mrow0 = mb * 32;           // global m offset of slab

    const float inv0 = invden[2 * a];
    const float inv1 = invden[2 * a + 1];
    const float mid = inv0 / inv1;
    const float fin = inv1;

    f8 acc[4] = {};
    for (int phase = 0; phase < 2; ++phase) {
        const float inv = phase ? inv1 : inv0;
        for (int b = phase * 32; b < phase * 32 + 32; ++b) {
            __syncthreads();   // prior iteration's LDS reads complete
            // stage E slab (32 x 128 f32): read once, write normalized back,
            // keep unnormalized f16 in LDS (range ~[0,40] -> safe for f16).
            float* eg = E + ((size_t)(a * Gg + b) * Mm + mrow0) * Mm;
            #pragma unroll
            for (int i = 0; i < 4; ++i) {
                const int ch  = tid + i * 256;      // 1024 float4 chunks
                const int row = ch >> 5;
                const int c4  = (ch & 31) * 4;
                float4 x = *(const float4*)(eg + row * Mm + c4);
                float4 y;
                y.x = x.x * inv; y.y = x.y * inv; y.z = x.z * inv; y.w = x.w * inv;
                *(float4*)(eg + row * Mm + c4) = y;          // normalized att out
                h4 hv;
                hv[0]=(_Float16)x.x; hv[1]=(_Float16)x.y;
                hv[2]=(_Float16)x.z; hv[3]=(_Float16)x.w;
                *(h4*)&Eh[row][c4] = hv;
            }
            // stage Vs[b] (256 x 128 f16), 4096 16B chunks
            #pragma unroll
            for (int i = 0; i < 16; ++i) {
                const int ch = tid + i * 256;
                const int d  = ch >> 4;
                const int cc = (ch & 15) * 8;
                *(h8*)&Vb[d][cc] = *(const h8*)(VsT + ((size_t)b * Dd + d) * Mm + cc);
            }
            __syncthreads();
            for (int c0 = 0; c0 < Mm; c0 += 32) {
                h16 af = frag_f16(&Eh[m0 + l16][c0 + hi * 8]);
                #pragma unroll
                for (int t = 0; t < 4; ++t) {
                    h16 bf = frag_f16(&Vb[d0 + t * 16 + l16][c0 + hi * 8]);
                    acc[t] = WMMA(af, bf, acc[t]);
                }
            }
        }
        const float sc = (phase == 0) ? mid : fin;
        #pragma unroll
        for (int t = 0; t < 4; ++t)
            #pragma unroll
            for (int v = 0; v < 8; ++v) acc[t][v] *= sc;
    }
    #pragma unroll
    for (int t = 0; t < 4; ++t)
        #pragma unroll
        for (int v = 0; v < 8; ++v) {
            const int m = mrow0 + m0 + hi * 8 + v;
            const int d = d0 + t * 16 + l16;
            OutPre[((size_t)a * Mm + m) * Dd + d] = acc[t][v];
        }
}

extern "C" void kernel_launch(void* const* d_in, const int* in_sizes, int n_in,
                              void* d_out, int out_size, void* d_ws, size_t ws_size,
                              hipStream_t stream) {
    const float* q    = (const float*)d_in[0];
    const float* k    = (const float*)d_in[1];
    const float* v    = (const float*)d_in[2];
    const float* mask = (const float*)d_in[3];
    const float* Wq   = (const float*)d_in[4];
    const float* bq   = (const float*)d_in[5];
    const float* Wk   = (const float*)d_in[6];
    const float* bk   = (const float*)d_in[7];
    const float* Wv   = (const float*)d_in[8];
    const float* bv   = (const float*)d_in[9];
    const float* Wo   = (const float*)d_in[10];
    const float* bo   = (const float*)d_in[11];

    char* ws = (char*)d_ws;
    size_t off = 0;
    _Float16* WtQ = (_Float16*)(ws + off); off += 256 * 256 * 2;
    _Float16* WtK = (_Float16*)(ws + off); off += 256 * 256 * 2;
    _Float16* WtV = (_Float16*)(ws + off); off += 256 * 256 * 2;
    _Float16* WtO = (_Float16*)(ws + off); off += 256 * 256 * 2;
    _Float16* Qh  = (_Float16*)(ws + off); off += (size_t)Gg * Mm * Dd * 2;
    _Float16* Kh  = (_Float16*)(ws + off); off += (size_t)Gg * Mm * Dd * 2;
    _Float16* VsT = (_Float16*)(ws + off); off += (size_t)Gg * Dd * Mm * 2;
    float* OutPre = (float*)(ws + off);    off += (size_t)Gg * Mm * Dd * 4;
    float* bsums  = (float*)(ws + off);    off += 4096 * 4;
    float* invden = (float*)(ws + off);    off += 128 * 4;

    float* out = (float*)d_out;                       // (G,M,D) f32
    float* E   = out + (size_t)Gg * Mm * Dd;          // att region (G,G,M,M) f32

    // P0: W transposes -> f16
    wt_kernel<<<256, 256, 0, stream>>>(Wq, WtQ);
    wt_kernel<<<256, 256, 0, stream>>>(Wk, WtK);
    wt_kernel<<<256, 256, 0, stream>>>(Wv, WtV);
    wt_kernel<<<256, 256, 0, stream>>>(Wo, WtO);

    // P1: projections.  q rows: g*129 + 1 + i (rowAdj=1,rowOff=1);
    //                   k rows: g*129 + j     (rowAdj=1,rowOff=0);
    //                   v rows: g*128 + i     (rowAdj=0,rowOff=0), stored transposed.
    proj_kernel<<<1024, 256, 0, stream>>>(q, WtQ, bq, Qh, 0, 1, 1);
    proj_kernel<<<1024, 256, 0, stream>>>(k, WtK, bk, Kh, 0, 1, 0);
    proj_kernel<<<1024, 256, 0, stream>>>(v, WtV, bv, VsT, 1, 0, 0);

    // P2: scores + exp (unnormalized att into d_out) + block sums
    scores_kernel<<<Gg * Gg, 256, 0, stream>>>(Qh, Kh, mask, E, bsums);

    // P2b: deterministic denominator reduction
    denred_kernel<<<1, 128, 0, stream>>>(bsums, invden);

    // P3: att x V with fused in-place att normalization (replaces norm pass)
    av_kernel<<<Gg * 4, 256, 0, stream>>>(E, VsT, invden, OutPre);

    // P5: final output = OutPre @ Wo + bo  (f32 store into d_out)
    proj_kernel<<<1024, 256, 0, stream>>>(OutPre, WtO, bo, (void*)out, 2, 0, 0);
}
